// Qwen2MoeSparseMoeBlockwithCache_29429115912761
// MI455X (gfx1250) — compile-verified, matrix operand order checked by reading, not run on hardware
//
#include <hip/hip_runtime.h>
#include <cstdint>

#define E_    16
#define TOPK_ 4
#define H_    2048
#define I_    1408
#define SI_   5632
#define CAP_  1024
#define T_    2048
#define NA_   (T_ * TOPK_)   // 8192 assignments

typedef __attribute__((ext_vector_type(16))) _Float16 v16h;
typedef __attribute__((ext_vector_type(8)))  _Float16 v8h_t;
typedef __attribute__((ext_vector_type(8)))  float    v8f;

// ---------------------------------------------------------------- utilities
__global__ void zero_u32(uint32_t* __restrict__ p, long n) {
    long i = (long)blockIdx.x * blockDim.x + threadIdx.x;
    if (i < n) p[i] = 0u;
}

__global__ void cvt_half(const float* __restrict__ x, _Float16* __restrict__ xh, long n) {
    long i = (long)blockIdx.x * blockDim.x + threadIdx.x;
    if (i < n) xh[i] = (_Float16)x[i];
}

// ------------------------------------------------------------------ router
__global__ __launch_bounds__(256) void router_kernel(
    const float* __restrict__ x, const float* __restrict__ gw,
    float* __restrict__ logits_out, int* __restrict__ fe, float* __restrict__ fw)
{
    const int t   = blockIdx.x;
    const int tid = threadIdx.x;
    const int e   = tid >> 4;      // expert 0..15
    const int j   = tid & 15;      // reducer lane 0..15
    const float* xr = x  + (long)t * H_;
    const float* wr = gw + (long)e * H_;
    float s = 0.f;
    for (int h = j * 4; h < H_; h += 64) {
        float4 xv = *(const float4*)(xr + h);
        float4 wv = *(const float4*)(wr + h);
        s += xv.x * wv.x + xv.y * wv.y + xv.z * wv.z + xv.w * wv.w;
    }
    __shared__ float red[256];
    __shared__ float lg[16];
    red[tid] = s;
    __syncthreads();
    for (int st = 8; st > 0; st >>= 1) {
        if (j < st) red[tid] += red[tid + st];
        __syncthreads();
    }
    if (j == 0) lg[e] = red[tid];
    __syncthreads();
    if (tid < 16) logits_out[t * E_ + tid] = lg[tid];
    if (tid == 0) {
        float mx = lg[0];
        for (int i = 1; i < E_; ++i) mx = fmaxf(mx, lg[i]);
        float p[E_]; float sum = 0.f;
        for (int i = 0; i < E_; ++i) { p[i] = __expf(lg[i] - mx); sum += p[i]; }
        float inv = __builtin_amdgcn_rcpf(sum);
        for (int i = 0; i < E_; ++i) p[i] *= inv;
        for (int k = 0; k < TOPK_; ++k) {
            int bi = 0; float bv = p[0];
            for (int i = 1; i < E_; ++i) if (p[i] > bv) { bv = p[i]; bi = i; }
            fe[t * TOPK_ + k] = bi;
            fw[t * TOPK_ + k] = bv;
            p[bi] = -1.f;
        }
    }
}

// -------------------------------------------- arrival-order expert positions
__global__ void hist_kernel(const int* __restrict__ fe, int* __restrict__ hist) {
    __shared__ int h[E_];
    if (threadIdx.x < E_) h[threadIdx.x] = 0;
    __syncthreads();
    int a = blockIdx.x * 256 + threadIdx.x;
    atomicAdd(&h[fe[a]], 1);
    __syncthreads();
    if (threadIdx.x < E_) hist[blockIdx.x * E_ + threadIdx.x] = h[threadIdx.x];
}

__global__ void prefix_kernel(const int* __restrict__ hist,
                              int* __restrict__ tbas, int* __restrict__ cnt) {
    int e = threadIdx.x;
    if (e >= E_) return;
    int run = 0;
    for (int t = 0; t < NA_ / 256; ++t) { tbas[t * E_ + e] = run; run += hist[t * E_ + e]; }
    cnt[e] = run > CAP_ ? CAP_ : run;
}

__global__ __launch_bounds__(256) void scatter_kernel(
    const int* __restrict__ fe, const float* __restrict__ fw,
    const int* __restrict__ tbas, int* __restrict__ tok, float* __restrict__ wgt)
{
    __shared__ int sfe[256];
    const int a0 = blockIdx.x * 256;
    const int i  = threadIdx.x;
    sfe[i] = fe[a0 + i];
    __syncthreads();
    const int e = sfe[i];
    int rank = 0;
    for (int q = 0; q < 256; ++q)
        if (q < i && sfe[q] == e) ++rank;
    int pos = tbas[blockIdx.x * E_ + e] + rank;
    if (pos < CAP_) {
        tok[e * CAP_ + pos] = (a0 + i) >> 2;   // token id (K=4)
        wgt[e * CAP_ + pos] = fw[a0 + i];
    }
}

// ------------------------------------------------------ shared-expert gate
__global__ __launch_bounds__(256) void sgate_kernel(
    const float* __restrict__ x, const float* __restrict__ sgw, float* __restrict__ sg)
{
    const int wave = threadIdx.x >> 5, lane = threadIdx.x & 31;
    const int t = blockIdx.x * 8 + wave;
    const float* xr = x + (long)t * H_;
    float s = 0.f;
    for (int h = lane * 4; h < H_; h += 128) {
        float4 xv = *(const float4*)(xr + h);
        float4 wv = *(const float4*)(sgw + h);
        s += xv.x * wv.x + xv.y * wv.y + xv.z * wv.z + xv.w * wv.w;
    }
    for (int off = 16; off > 0; off >>= 1) s += __shfl_down(s, off, 32);
    if (lane == 0) sg[t] = __builtin_amdgcn_rcpf(1.f + __expf(-s));
}

// -------------------------------------------------------------- WMMA GEMM
// B fragment: lane l holds column n=l&15; K pairs per VGPR group (fp32->f16 cvt)
__device__ __forceinline__ void load_bfrag(const float* __restrict__ row, int k, int hs8, v16h& bf) {
    const float4 f0 = *(const float4*)(row + k + hs8);
    const float4 f1 = *(const float4*)(row + k + hs8 + 4);
    const float4 g0 = *(const float4*)(row + k + hs8 + 16);
    const float4 g1 = *(const float4*)(row + k + hs8 + 20);
    bf[0]=(_Float16)f0.x;  bf[1]=(_Float16)f0.y;  bf[2]=(_Float16)f0.z;  bf[3]=(_Float16)f0.w;
    bf[4]=(_Float16)f1.x;  bf[5]=(_Float16)f1.y;  bf[6]=(_Float16)f1.z;  bf[7]=(_Float16)f1.w;
    bf[8]=(_Float16)g0.x;  bf[9]=(_Float16)g0.y;  bf[10]=(_Float16)g0.z; bf[11]=(_Float16)g0.w;
    bf[12]=(_Float16)g1.x; bf[13]=(_Float16)g1.y; bf[14]=(_Float16)g1.z; bf[15]=(_Float16)g1.w;
}

// Block tile: 64 M x 128 N, 8 waves; each wave: one 16-col strip, 4 M-subtiles.
// A tile staged via CDNA5 async copy (global_load_async_to_lds_b128, ASYNCcnt),
// double-buffered in LDS so the k+32 copy overlaps the k-step WMMAs.
// MODE 0: C(half)[m,n] = silu(A@B0^T) * (A@B1^T)           (gate/up fusion)
// MODE 1: atomicAdd(Cout[row(m), n], scale(m) * (A@B0^T))  (down + combine)
template<int MODE>
__global__ __launch_bounds__(256) void wmma_gemm(
    const _Float16* __restrict__ A, long aOffPerZ, int ldA, int Kd,
    const float* __restrict__ B0, const float* __restrict__ B1, long bOffPerZ,
    void* __restrict__ Cout, long cOffPerZ, int ldC,
    const int* __restrict__ rowIdxA,        // per-z stride CAP_; A row gather
    const int* __restrict__ rowIdxOut,      // per-z stride CAP_; out row scatter
    const float* __restrict__ rowScale,     // per-z stride CAP_; combine weight
    const int* __restrict__ countArr, int Mfixed)
{
    const int z = blockIdx.z;
    int Mcnt = countArr ? countArr[z] : Mfixed;
    if (countArr && Mcnt > CAP_) Mcnt = CAP_;
    const int m0 = blockIdx.y * 64;
    if (m0 >= Mcnt) return;

    const int tid  = threadIdx.x;
    const int lane = tid & 31;
    const int wave = tid >> 5;
    const int nout = blockIdx.x * 128 + wave * 16 + (lane & 15);
    const int hs8  = (lane >> 4) * 8;
    const int mf   = lane & 15;

    const _Float16* Az = A + (long)z * aOffPerZ;
    const int* ridxA = rowIdxA ? rowIdxA + (long)z * CAP_ : nullptr;
    const float* B0row = B0 + (long)z * bOffPerZ + (long)nout * Kd;
    const float* B1row = (MODE == 0) ? (B1 + (long)z * bOffPerZ + (long)nout * Kd) : B0;

    // cooperative A staging: thread -> row tid>>2 (0..63), col group (tid&3)*8
    const int arl = tid >> 2;
    const int acg = (tid & 3) * 8;
    const long arow = ridxA ? (long)ridxA[m0 + arl] : (long)(m0 + arl);
    const _Float16* Asrc = Az + arow * ldA + acg;

    __shared__ _Float16 As[2][64][40];   // 80B row pitch (16B-aligned), double buffer
    const uint32_t ldsWr0 = (uint32_t)(uintptr_t)(&As[0][arl][acg]);
    const uint32_t ldsWr1 = (uint32_t)(uintptr_t)(&As[1][arl][acg]);

    v8f accg[4] = {{0}, {0}, {0}, {0}};
    v8f accu[4] = {{0}, {0}, {0}, {0}};

    // prologue: kick off tile 0
    asm volatile("global_load_async_to_lds_b128 %0, %1, off"
                 :: "v"(ldsWr0), "v"(Asrc) : "memory");

    for (int k = 0, buf = 0; k < Kd; k += 32, buf ^= 1) {
        asm volatile("s_wait_asynccnt 0" ::: "memory");
        __syncthreads();                       // tile `buf` visible to all waves
        if (k + 32 < Kd) {                     // overlap next A tile with compute
            asm volatile("global_load_async_to_lds_b128 %0, %1, off"
                         :: "v"(buf ? ldsWr0 : ldsWr1), "v"(Asrc + k + 32) : "memory");
        }
        __builtin_prefetch(B0row + k + 32, 0, 1);          // global_prefetch_b8
        if (MODE == 0) __builtin_prefetch(B1row + k + 32, 0, 1);

        v16h bf0, bf1;
        load_bfrag(B0row, k, hs8, bf0);
        if (MODE == 0) load_bfrag(B1row, k, hs8, bf1);

        #pragma unroll
        for (int t = 0; t < 4; ++t) {
            v16h af;
            {
                v8h_t lo = *(const v8h_t*)(&As[buf][mf + 16 * t][hs8]);        // K=hs8..+7
                v8h_t hi = *(const v8h_t*)(&As[buf][mf + 16 * t][16 + hs8]);   // K=16+hs8..
                #pragma unroll
                for (int i = 0; i < 8; ++i) { af[i] = lo[i]; af[8 + i] = hi[i]; }
            }
            accg[t] = __builtin_amdgcn_wmma_f32_16x16x32_f16(false, af, false, bf0,
                                                             (short)0, accg[t], false, false);
            if (MODE == 0)
                accu[t] = __builtin_amdgcn_wmma_f32_16x16x32_f16(false, af, false, bf1,
                                                                 (short)0, accu[t], false, false);
        }
        __syncthreads();                       // all reads of `buf` done before rewrite
    }

    const int mh = (lane >> 4) * 8;    // C/D: vgpr r -> row r + 8*(lane>>4)
    if (MODE == 0) {
        _Float16* C = (_Float16*)Cout + (long)z * cOffPerZ;
        #pragma unroll
        for (int t = 0; t < 4; ++t) {
            #pragma unroll
            for (int r = 0; r < 8; ++r) {
                const int m = m0 + 16 * t + mh + r;
                float g = accg[t][r];
                float sig = __builtin_amdgcn_rcpf(1.f + __expf(-g));  // v_exp + v_rcp
                C[(long)m * ldC + nout] = (_Float16)(g * sig * accu[t][r]);
            }
        }
    } else {
        float* C = (float*)Cout;
        const int*   ridxO = rowIdxOut ? rowIdxOut + (long)z * CAP_ : nullptr;
        const float* rscl  = rowScale  ? rowScale  + (long)z * CAP_ : nullptr;
        #pragma unroll
        for (int t = 0; t < 4; ++t) {
            #pragma unroll
            for (int r = 0; r < 8; ++r) {
                const int lm = m0 + 16 * t + mh + r;
                const long orow = ridxO ? (long)ridxO[lm] : (long)lm;
                const float s = rscl ? rscl[lm] : 1.f;
                atomicAdd(&C[orow * (long)ldC + nout], s * accg[t][r]);
            }
        }
    }
}

// ---------------------------------------------------------------- launcher
extern "C" void kernel_launch(void* const* d_in, const int* in_sizes, int n_in,
                              void* d_out, int out_size, void* d_ws, size_t ws_size,
                              hipStream_t stream)
{
    (void)in_sizes; (void)n_in; (void)out_size; (void)ws_size;
    const float* x       = (const float*)d_in[0];
    const float* gate_w  = (const float*)d_in[1];
    const float* we_gate = (const float*)d_in[2];
    const float* we_up   = (const float*)d_in[3];
    const float* we_down = (const float*)d_in[4];
    const float* wsg     = (const float*)d_in[5];
    const float* wsu     = (const float*)d_in[6];
    const float* wsd     = (const float*)d_in[7];
    const float* sg_w    = (const float*)d_in[8];
    float* out = (float*)d_out;                     // [T*H] tokens, then [T*E] logits

    char* p = (char*)d_ws;
    auto alloc = [&](size_t bytes) { char* r = p; p += (bytes + 255) & ~(size_t)255; return r; };
    _Float16* xh  = (_Float16*)alloc((size_t)T_ * H_ * 2);        // f16 activations
    _Float16* act = (_Float16*)alloc((size_t)E_ * CAP_ * I_ * 2); // expert intermediates
    _Float16* hsb = (_Float16*)alloc((size_t)T_ * SI_ * 2);       // shared intermediates
    int*   fe   = (int*)  alloc((size_t)NA_ * 4);
    float* fw   = (float*)alloc((size_t)NA_ * 4);
    int*   hist = (int*)  alloc((size_t)(NA_ / 256) * E_ * 4);
    int*   tbas = (int*)  alloc((size_t)(NA_ / 256) * E_ * 4);
    int*   cnt  = (int*)  alloc((size_t)E_ * 4);
    int*   tok  = (int*)  alloc((size_t)E_ * CAP_ * 4);
    float* wgt  = (float*)alloc((size_t)E_ * CAP_ * 4);
    float* sg   = (float*)alloc((size_t)T_ * 4);

    // deterministic across graph replays: zero accumulation targets each call
    zero_u32<<<(T_ * H_ + 255) / 256, 256, 0, stream>>>((uint32_t*)out, (long)T_ * H_);
    zero_u32<<<(E_ * CAP_ + 255) / 256, 256, 0, stream>>>((uint32_t*)tok, (long)E_ * CAP_);
    zero_u32<<<(E_ * CAP_ + 255) / 256, 256, 0, stream>>>((uint32_t*)wgt, (long)E_ * CAP_);

    router_kernel<<<T_, 256, 0, stream>>>(x, gate_w, out + (size_t)T_ * H_, fe, fw);
    hist_kernel<<<NA_ / 256, 256, 0, stream>>>(fe, hist);
    prefix_kernel<<<1, 16, 0, stream>>>(hist, tbas, cnt);
    scatter_kernel<<<NA_ / 256, 256, 0, stream>>>(fe, fw, tbas, tok, wgt);
    cvt_half<<<(T_ * H_) / 256, 256, 0, stream>>>(x, xh, (long)T_ * H_);
    sgate_kernel<<<T_ / 8, 256, 0, stream>>>(x, sg_w, sg);

    // experts: gate/up  (A rows gathered by token id; silu(g)*u -> act)
    wmma_gemm<0><<<dim3(I_ / 128, CAP_ / 64, E_), 256, 0, stream>>>(
        xh, 0L, H_, H_, we_gate, we_up, (long)I_ * H_,
        act, (long)CAP_ * I_, I_, tok, nullptr, nullptr, cnt, 0);
    // experts: down + weighted scatter-add into out
    wmma_gemm<1><<<dim3(H_ / 128, CAP_ / 64, E_), 256, 0, stream>>>(
        act, (long)CAP_ * I_, I_, I_, we_down, we_down, (long)H_ * I_,
        out, 0L, H_, nullptr, tok, wgt, cnt, 0);
    // shared expert: gate/up
    wmma_gemm<0><<<dim3(SI_ / 128, T_ / 64, 1), 256, 0, stream>>>(
        xh, 0L, H_, H_, wsg, wsu, 0L,
        hsb, 0L, SI_, nullptr, nullptr, nullptr, nullptr, T_);
    // shared expert: down, scaled by sigmoid gate, add into out
    wmma_gemm<1><<<dim3(H_ / 128, T_ / 64, 1), 256, 0, stream>>>(
        hsb, 0L, SI_, SI_, wsd, wsd, 0L,
        out, 0L, H_, nullptr, nullptr, sg, nullptr, T_);
}
